// TokenUOTBlock_89386859364644
// MI455X (gfx1250) — matrix operand order repaired and checked by reading, not compiled, not run on hardware
//
#include <hip/hip_runtime.h>

// ---------------------------------------------------------------------------
// Types for CDNA5 WMMA
// ---------------------------------------------------------------------------
typedef __attribute__((ext_vector_type(16))) __bf16 bf16x16;
typedef __attribute__((ext_vector_type(8)))  __bf16 bf16x8;
typedef __attribute__((ext_vector_type(8)))  float  f32x8;

#define WMMA_BF16(a, b, c) \
  __builtin_amdgcn_wmma_f32_16x16x32_bf16(false, (a), false, (b), (short)0, (c), false, false)

// Problem constants
#define BATCH   2
#define NTOK    1024
#define NHEAD   16
#define DHEAD   32
#define DMODEL  512
#define DFF     2048
#define BN      (BATCH * NTOK)          // 2048
#define BH      (BATCH * NHEAD)         // 32
#define BHN     (BATCH * NHEAD * NTOK)  // 32768

// ---------------------------------------------------------------------------
// Fragment loaders (16-bit operands, wave32)
// A-matrix 16x32: lanes 0-15 row M=lane, K {0..7, 16..23}; lanes 16-31 K {8..15, 24..31}
// B-matrix 32x16 (loaded from BT, N-major): lanes 0-15 col N=lane, K {0..7, 16..23}; lanes 16-31 K {8..15, 24..31}
// ---------------------------------------------------------------------------
__device__ __forceinline__ bf16x16 load_fragA(const __bf16* p0, int ld, int lane) {
  int r = lane & 15, half = lane >> 4;
  const __bf16* p = p0 + (size_t)r * ld + half * 8;
  bf16x8 lo = *(const bf16x8*)p;
  bf16x8 hi = *(const bf16x8*)(p + 16);
  bf16x16 o;
#pragma unroll
  for (int i = 0; i < 8; ++i) { o[i] = lo[i]; o[i + 8] = hi[i]; }
  return o;
}

__device__ __forceinline__ bf16x16 load_fragB(const __bf16* p0, int ld, int lane) {
  int r = lane & 15, half = lane >> 4;
  const __bf16* p = p0 + (size_t)r * ld + half * 8;
  bf16x8 lo = *(const bf16x8*)p;
  bf16x8 hi = *(const bf16x8*)(p + 16);
  bf16x16 o;
#pragma unroll
  for (int i = 0; i < 8; ++i) { o[i] = lo[i]; o[i + 8] = hi[i]; }
  return o;
}

__device__ __forceinline__ float wave_sum(float x) {
#pragma unroll
  for (int off = 16; off >= 1; off >>= 1) x += __shfl_xor(x, off, 32);
  return x;
}

// ---------------------------------------------------------------------------
// Weight prep: fp32 W[K,N] -> bf16 WT[N,K]
// ---------------------------------------------------------------------------
__global__ void wprep_kernel(const float* __restrict__ W, __bf16* __restrict__ WT,
                             int K, int N) {
  int idx = blockIdx.x * blockDim.x + threadIdx.x;
  if (idx >= K * N) return;
  int k = idx / N, n = idx % N;
  WT[(size_t)n * K + k] = (__bf16)W[idx];
}

// ---------------------------------------------------------------------------
// LayerNorm over 512 cols with scale/bias, bf16 output. One wave per row.
// ---------------------------------------------------------------------------
__global__ void ln512_kernel(const float* __restrict__ X, const float* __restrict__ scale,
                             const float* __restrict__ bias, __bf16* __restrict__ Y) {
  int w = (blockIdx.x * blockDim.x + threadIdx.x) >> 5;
  int lane = threadIdx.x & 31;
  const float* x = X + (size_t)w * DMODEL;
  float v[16];
  float s = 0.f, ss = 0.f;
#pragma unroll
  for (int t = 0; t < 16; ++t) {
    float a = x[lane + 32 * t];
    v[t] = a; s += a; ss += a * a;
  }
  s = wave_sum(s); ss = wave_sum(ss);
  float mean = s * (1.f / DMODEL);
  float var  = ss * (1.f / DMODEL) - mean * mean;
  float inv  = rsqrtf(var + 1e-5f);
#pragma unroll
  for (int t = 0; t < 16; ++t) {
    int c = lane + 32 * t;
    Y[(size_t)w * DMODEL + c] = (__bf16)((v[t] - mean) * inv * scale[c] + bias[c]);
  }
}

// ---------------------------------------------------------------------------
// Generic bf16 GEMM: C[M,N] = A[M,K] @ BT[N,K]^T (+ optional fp32 residual R)
// block = 256 threads = 8 waves (4 along M, 2 along N); wave tile 32x32
// ---------------------------------------------------------------------------
__global__ __launch_bounds__(256)
void gemm_bf16_kernel(const __bf16* __restrict__ A, const __bf16* __restrict__ BT,
                      const float* __restrict__ R, float* __restrict__ C,
                      int M, int N, int K) {
  int lane = threadIdx.x & 31;
  int wid  = threadIdx.x >> 5;
  int m0 = blockIdx.y * 128 + (wid & 3) * 32;
  int n0 = blockIdx.x * 64 + (wid >> 2) * 32;
  f32x8 acc00 = {}, acc01 = {}, acc10 = {}, acc11 = {};
  for (int k0 = 0; k0 < K; k0 += 32) {
    bf16x16 a0 = load_fragA(A + (size_t)m0 * K + k0, K, lane);
    bf16x16 a1 = load_fragA(A + (size_t)(m0 + 16) * K + k0, K, lane);
    bf16x16 b0 = load_fragB(BT + (size_t)n0 * K + k0, K, lane);
    bf16x16 b1 = load_fragB(BT + (size_t)(n0 + 16) * K + k0, K, lane);
    acc00 = WMMA_BF16(a0, b0, acc00);
    acc01 = WMMA_BF16(a0, b1, acc01);
    acc10 = WMMA_BF16(a1, b0, acc10);
    acc11 = WMMA_BF16(a1, b1, acc11);
  }
  int r = lane & 15, half = lane >> 4;
  const f32x8 accs[4] = {acc00, acc01, acc10, acc11};
#pragma unroll
  for (int t = 0; t < 4; ++t) {
    int mt = m0 + (t >> 1) * 16, nt = n0 + (t & 1) * 16;
#pragma unroll
    for (int rr = 0; rr < 8; ++rr) {
      size_t idx = (size_t)(mt + half * 8 + rr) * N + nt + r;
      float v = accs[t][rr];
      if (R) v += R[idx];
      C[idx] = v;
    }
  }
}

// ---------------------------------------------------------------------------
// Fused FFN up: g = silu(A@W1T^T) * (A@W2T^T), bf16 output
// ---------------------------------------------------------------------------
__global__ __launch_bounds__(256)
void ffn_dual_kernel(const __bf16* __restrict__ A, const __bf16* __restrict__ B1T,
                     const __bf16* __restrict__ B2T, __bf16* __restrict__ G,
                     int M, int N, int K) {
  int lane = threadIdx.x & 31;
  int wid  = threadIdx.x >> 5;
  int m0 = blockIdx.y * 128 + (wid & 3) * 32;
  int n0 = blockIdx.x * 64 + (wid >> 2) * 32;
  f32x8 p00 = {}, p01 = {}, p10 = {}, p11 = {};
  f32x8 q00 = {}, q01 = {}, q10 = {}, q11 = {};
  for (int k0 = 0; k0 < K; k0 += 32) {
    bf16x16 a0 = load_fragA(A + (size_t)m0 * K + k0, K, lane);
    bf16x16 a1 = load_fragA(A + (size_t)(m0 + 16) * K + k0, K, lane);
    bf16x16 c0 = load_fragB(B1T + (size_t)n0 * K + k0, K, lane);
    bf16x16 c1 = load_fragB(B1T + (size_t)(n0 + 16) * K + k0, K, lane);
    bf16x16 d0 = load_fragB(B2T + (size_t)n0 * K + k0, K, lane);
    bf16x16 d1 = load_fragB(B2T + (size_t)(n0 + 16) * K + k0, K, lane);
    p00 = WMMA_BF16(a0, c0, p00); p01 = WMMA_BF16(a0, c1, p01);
    p10 = WMMA_BF16(a1, c0, p10); p11 = WMMA_BF16(a1, c1, p11);
    q00 = WMMA_BF16(a0, d0, q00); q01 = WMMA_BF16(a0, d1, q01);
    q10 = WMMA_BF16(a1, d0, q10); q11 = WMMA_BF16(a1, d1, q11);
  }
  int r = lane & 15, half = lane >> 4;
  const f32x8 ps[4] = {p00, p01, p10, p11};
  const f32x8 qs[4] = {q00, q01, q10, q11};
#pragma unroll
  for (int t = 0; t < 4; ++t) {
    int mt = m0 + (t >> 1) * 16, nt = n0 + (t & 1) * 16;
#pragma unroll
    for (int rr = 0; rr < 8; ++rr) {
      size_t idx = (size_t)(mt + half * 8 + rr) * N + nt + r;
      float u1 = ps[t][rr], u2 = qs[t][rr];
      float s = u1 / (1.f + __expf(-u1));   // silu
      G[idx] = (__bf16)(s * u2);
    }
  }
}

// ---------------------------------------------------------------------------
// Per-head prep: LN(Q),LN(K) over DHEAD -> bf16 head-major; V -> fp32 head-major
// One wave per (b,h,n)
// ---------------------------------------------------------------------------
__global__ void headprep_kernel(const float* __restrict__ Qf, const float* __restrict__ Kf,
                                const float* __restrict__ Vf, __bf16* __restrict__ Qln,
                                __bf16* __restrict__ Kln, float* __restrict__ Vh) {
  int w = (blockIdx.x * blockDim.x + threadIdx.x) >> 5;   // (bh, n)
  int lane = threadIdx.x & 31;
  int bh = w >> 10, n = w & 1023;
  int b = bh >> 4, h = bh & 15;
  size_t src = ((size_t)b * NTOK + n) * DMODEL + h * DHEAD + lane;
  size_t dst = (size_t)w * DHEAD + lane;
  float q = Qf[src], k = Kf[src];
  Vh[dst] = Vf[src];
  float mq = wave_sum(q) * (1.f / DHEAD);
  float dq = q - mq;
  float vq = wave_sum(dq * dq) * (1.f / DHEAD);
  Qln[dst] = (__bf16)(dq * rsqrtf(vq + 1e-5f));
  float mk = wave_sum(k) * (1.f / DHEAD);
  float dk = k - mk;
  float vk = wave_sum(dk * dk) * (1.f / DHEAD);
  Kln[dst] = (__bf16)(dk * rsqrtf(vk + 1e-5f));
}

// ---------------------------------------------------------------------------
// Cost/log-kernel: LK = -(content + w_rel + geo_bias)/eps[h]
// content tile via single WMMA (K = DHEAD = 32). Epilogue fuses cdist + bias.
// ---------------------------------------------------------------------------
__global__ __launch_bounds__(256)
void content_kernel(const __bf16* __restrict__ Qln, const __bf16* __restrict__ Kln,
                    const float* __restrict__ wrel, const float* __restrict__ xres,
                    const float* __restrict__ wdist, const float* __restrict__ eps,
                    float* __restrict__ LK) {
  int lane = threadIdx.x & 31;
  int wid  = threadIdx.x >> 5;
  int bh = blockIdx.z, b = bh >> 4, h = bh & 15;
  int m0 = blockIdx.y * 128 + (wid & 3) * 32;   // i
  int n0 = blockIdx.x * 64 + (wid >> 2) * 32;   // j
  const __bf16* Ab = Qln + ((size_t)bh * NTOK + m0) * DHEAD;
  const __bf16* Bb = Kln + ((size_t)bh * NTOK + n0) * DHEAD;
  bf16x16 a0 = load_fragA(Ab, DHEAD, lane);
  bf16x16 a1 = load_fragA(Ab + 16 * DHEAD, DHEAD, lane);
  bf16x16 b0 = load_fragB(Bb, DHEAD, lane);
  bf16x16 b1 = load_fragB(Bb + 16 * DHEAD, DHEAD, lane);
  f32x8 z = {};
  f32x8 acc00 = WMMA_BF16(a0, b0, z);
  f32x8 acc01 = WMMA_BF16(a0, b1, z);
  f32x8 acc10 = WMMA_BF16(a1, b0, z);
  f32x8 acc11 = WMMA_BF16(a1, b1, z);
  int r = lane & 15, half = lane >> 4;
  float wd  = wdist[h];
  float nie = -1.f / eps[h];
  const float* Xb = xres + (size_t)b * NTOK * 3;
  const float* wr = wrel + (size_t)bh * NTOK * NTOK;
  float* lk = LK + (size_t)bh * NTOK * NTOK;
  const f32x8 accs[4] = {acc00, acc01, acc10, acc11};
#pragma unroll
  for (int t = 0; t < 4; ++t) {
    int mt = m0 + (t >> 1) * 16, nt = n0 + (t & 1) * 16;
    int j = nt + r;
    float xj0 = Xb[j * 3], xj1 = Xb[j * 3 + 1], xj2 = Xb[j * 3 + 2];
#pragma unroll
    for (int rr = 0; rr < 8; ++rr) {
      int i = mt + half * 8 + rr;
      float d0 = Xb[i * 3] - xj0, d1 = Xb[i * 3 + 1] - xj1, d2 = Xb[i * 3 + 2] - xj2;
      float dist = sqrtf(fmaxf(d0 * d0 + d1 * d1 + d2 * d2, 1e-12f));
      float fd = dist / (10.0f + dist);
      float Cv = -0.17677669529663687f * accs[t][rr]   // -(1/sqrt(32)) * QK
               + wr[(size_t)i * NTOK + j] + wd * fd;
      lk[(size_t)i * NTOK + j] = nie * Cv;
    }
  }
}

// ---------------------------------------------------------------------------
// Sinkhorn: row logsumexp update (log_u). One wave per row.
// ---------------------------------------------------------------------------
__global__ void sinkhorn_u_kernel(const float* __restrict__ LK, const float* __restrict__ log_v,
                                  const float* __restrict__ mu, const float* __restrict__ eps,
                                  float* __restrict__ log_u) {
  int w = (blockIdx.x * blockDim.x + threadIdx.x) >> 5;
  int lane = threadIdx.x & 31;
  int bh = w >> 10, h = bh & 15;
  const float* row = LK + (size_t)w * NTOK;
  const float* lv = log_v + (size_t)bh * NTOK;
  float m = -1e30f, s = 0.f;
#pragma unroll 4
  for (int t = 0; t < 32; ++t) {
    int j = lane + (t << 5);
    float x = row[j] + lv[j];
    float nm = fmaxf(m, x);
    s = s * __expf(m - nm) + __expf(x - nm);
    m = nm;
  }
#pragma unroll
  for (int off = 16; off >= 1; off >>= 1) {
    float om = __shfl_xor(m, off, 32);
    float os = __shfl_xor(s, off, 32);
    float nm = fmaxf(m, om);
    s = s * __expf(m - nm) + os * __expf(om - nm);
    m = nm;
  }
  if (lane == 0) {
    float fi = 1.f / (1.f + eps[h]);
    log_u[w] = fi * (__logf(fmaxf(mu[w], 1e-8f)) - (m + __logf(s)));
  }
}

// ---------------------------------------------------------------------------
// Sinkhorn: column logsumexp update (log_v). One 1024-thread block per (b,h).
// ---------------------------------------------------------------------------
__global__ __launch_bounds__(1024)
void sinkhorn_v_kernel(const float* __restrict__ LK, const float* __restrict__ log_u,
                       const float* __restrict__ nu, const float* __restrict__ eps,
                       float* __restrict__ log_v) {
  __shared__ float slu[NTOK];
  int bh = blockIdx.x, h = bh & 15, j = threadIdx.x;
  slu[j] = log_u[bh * NTOK + j];
  __syncthreads();
  const float* base = LK + (size_t)bh * NTOK * NTOK + j;
  float m = -1e30f, s = 0.f;
  for (int i = 0; i < NTOK; ++i) {
    float x = base[(size_t)i * NTOK] + slu[i];
    float nm = fmaxf(m, x);
    s = s * __expf(m - nm) + __expf(x - nm);
    m = nm;
  }
  float fi = 1.f / (1.f + eps[h]);
  log_v[bh * NTOK + j] = fi * (__logf(fmaxf(nu[bh * NTOK + j], 1e-8f)) - (m + __logf(s)));
}

// ---------------------------------------------------------------------------
// Transport: o = sigmoid(G) * (T_norm @ V); centroid = T_norm @ x_res
// One wave per (b,h,i); lane owns output channel d.
// ---------------------------------------------------------------------------
__global__ void transport_kernel(const float* __restrict__ LK, const float* __restrict__ log_u,
                                 const float* __restrict__ log_v, const float* __restrict__ Vh,
                                 const float* __restrict__ Gf, const float* __restrict__ xres,
                                 __bf16* __restrict__ o_bf, float* __restrict__ xcent) {
  int w = (blockIdx.x * blockDim.x + threadIdx.x) >> 5;
  int lane = threadIdx.x & 31;
  int bh = w >> 10, i = w & 1023;
  int b = bh >> 4, h = bh & 15;
  const float* row = LK + (size_t)w * NTOK;
  const float* lv = log_v + (size_t)bh * NTOK;
  const float* Vb = Vh + (size_t)bh * NTOK * DHEAD;
  const float* Xb = xres + (size_t)b * NTOK * 3;
  float lu = log_u[w];
  float oacc = 0.f, c0 = 0.f, c1 = 0.f, c2 = 0.f, ssum = 0.f;
  for (int chunk = 0; chunk < NTOK; chunk += 32) {
    int jl = chunk + lane;
    float tl = __expf(lu + row[jl] + lv[jl]);
    ssum += tl;
    float xl0 = Xb[jl * 3], xl1 = Xb[jl * 3 + 1], xl2 = Xb[jl * 3 + 2];
#pragma unroll
    for (int jj = 0; jj < 32; ++jj) {
      float tj = __shfl(tl, jj, 32);
      oacc += tj * Vb[(size_t)(chunk + jj) * DHEAD + lane];
      c0 += tj * __shfl(xl0, jj, 32);
      c1 += tj * __shfl(xl1, jj, 32);
      c2 += tj * __shfl(xl2, jj, 32);
    }
  }
  ssum = wave_sum(ssum);
  float inv = 1.f / (ssum + 1e-8f);
  size_t oidx = ((size_t)b * NTOK + i) * DMODEL + h * DHEAD + lane;
  float g = Gf[oidx];
  float sg = 1.f / (1.f + __expf(-g));
  o_bf[oidx] = (__bf16)(sg * oacc * inv);
  if (lane < 3) {
    float c = (lane == 0) ? c0 : ((lane == 1) ? c1 : c2);
    xcent[(size_t)w * 3 + lane] = c * inv;
  }
}

// ---------------------------------------------------------------------------
// x_out = x_res + sum_h gamma[h] * (x_res - centroid[h])
// ---------------------------------------------------------------------------
__global__ void xout_kernel(const float* __restrict__ xres, const float* __restrict__ gamma,
                            const float* __restrict__ xcent, float* __restrict__ out) {
  int idx = blockIdx.x * blockDim.x + threadIdx.x;
  if (idx >= BN * 3) return;
  int c = idx % 3, bn = idx / 3;
  int b = bn >> 10, n = bn & 1023;
  float gs = 0.f, acc = 0.f;
#pragma unroll
  for (int h = 0; h < NHEAD; ++h) {
    float gm = gamma[h];
    gs += gm;
    acc += gm * xcent[(((size_t)(b * NHEAD + h)) * NTOK + n) * 3 + c];
  }
  float xr = xres[idx];
  out[idx] = xr + gs * xr - acc;
}

__global__ void zero_kernel(float* p, int n) {
  int i = blockIdx.x * blockDim.x + threadIdx.x;
  if (i < n) p[i] = 0.f;
}

// ---------------------------------------------------------------------------
// Host-side orchestration
// ---------------------------------------------------------------------------
extern "C" void kernel_launch(void* const* d_in, const int* in_sizes, int n_in,
                              void* d_out, int out_size, void* d_ws, size_t ws_size,
                              hipStream_t stream) {
  (void)in_sizes; (void)n_in; (void)out_size; (void)ws_size;
  const float* h     = (const float*)d_in[0];
  const float* x_res = (const float*)d_in[1];
  const float* mu    = (const float*)d_in[2];
  const float* nu    = (const float*)d_in[3];
  const float* wrel  = (const float*)d_in[4];
  const float* wdist = (const float*)d_in[5];
  /* d_in[6] = pos_bins, unused by the reference */
  const float* ln_as = (const float*)d_in[7];
  const float* ln_ab = (const float*)d_in[8];
  const float* wq    = (const float*)d_in[9];
  const float* wk    = (const float*)d_in[10];
  const float* wv    = (const float*)d_in[11];
  const float* wg    = (const float*)d_in[12];
  const float* wo    = (const float*)d_in[13];
  const float* gamma = (const float*)d_in[14];
  const float* eps   = (const float*)d_in[15];
  const float* ln_fs = (const float*)d_in[16];
  const float* ln_fb = (const float*)d_in[17];
  const float* w1    = (const float*)d_in[18];
  const float* w2    = (const float*)d_in[19];
  const float* w3    = (const float*)d_in[20];

  // workspace carving
  char* base = (char*)d_ws;
  size_t off = 0;
  auto carve = [&](size_t bytes) -> void* {
    off = (off + 255) & ~(size_t)255;
    void* p = base + off;
    off += bytes;
    return p;
  };
  __bf16* hn_bf  = (__bf16*)carve((size_t)BN * DMODEL * 2);
  __bf16* wqT    = (__bf16*)carve((size_t)DMODEL * DMODEL * 2);
  __bf16* wkT    = (__bf16*)carve((size_t)DMODEL * DMODEL * 2);
  __bf16* wvT    = (__bf16*)carve((size_t)DMODEL * DMODEL * 2);
  __bf16* wgT    = (__bf16*)carve((size_t)DMODEL * DMODEL * 2);
  __bf16* woT    = (__bf16*)carve((size_t)DMODEL * DMODEL * 2);
  __bf16* w1T    = (__bf16*)carve((size_t)DFF * DMODEL * 2);
  __bf16* w2T    = (__bf16*)carve((size_t)DFF * DMODEL * 2);
  __bf16* w3T    = (__bf16*)carve((size_t)DMODEL * DFF * 2);
  float*  Qf     = (float*)carve((size_t)BN * DMODEL * 4);
  float*  Kf     = (float*)carve((size_t)BN * DMODEL * 4);
  float*  Vf     = (float*)carve((size_t)BN * DMODEL * 4);
  float*  Gf     = (float*)carve((size_t)BN * DMODEL * 4);
  __bf16* Qln    = (__bf16*)carve((size_t)BHN * DHEAD * 2);
  __bf16* Kln    = (__bf16*)carve((size_t)BHN * DHEAD * 2);
  float*  Vh     = (float*)carve((size_t)BHN * DHEAD * 4);
  float*  LK     = (float*)carve((size_t)BH * NTOK * NTOK * 4);   // 128 MB
  float*  log_u  = (float*)carve((size_t)BHN * 4);
  float*  log_v  = (float*)carve((size_t)BHN * 4);
  float*  xcent  = (float*)carve((size_t)BHN * 3 * 4);
  __bf16* o_bf   = (__bf16*)carve((size_t)BN * DMODEL * 2);
  float*  h2     = (float*)carve((size_t)BN * DMODEL * 4);
  __bf16* hff_bf = (__bf16*)carve((size_t)BN * DMODEL * 2);
  __bf16* g_bf   = (__bf16*)carve((size_t)BN * DFF * 2);

  float* out_h  = (float*)d_out;                 // (B,N,512)
  float* out_x  = out_h + (size_t)BN * DMODEL;   // (B,N,3)
  float* out_lu = out_x + (size_t)BN * 3;        // (B,H,N)
  float* out_lv = out_lu + (size_t)BHN;          // (B,H,N)

  // 1. weight transpose + bf16 convert
  auto wp = [&](const float* W, __bf16* WT, int K, int N) {
    int total = K * N;
    wprep_kernel<<<(total + 255) / 256, 256, 0, stream>>>(W, WT, K, N);
  };
  wp(wq, wqT, DMODEL, DMODEL); wp(wk, wkT, DMODEL, DMODEL);
  wp(wv, wvT, DMODEL, DMODEL); wp(wg, wgT, DMODEL, DMODEL);
  wp(wo, woT, DMODEL, DMODEL);
  wp(w1, w1T, DMODEL, DFF); wp(w2, w2T, DMODEL, DFF);
  wp(w3, w3T, DFF, DMODEL);

  // 2. attention layernorm -> bf16
  ln512_kernel<<<BN / 8, 256, 0, stream>>>(h, ln_as, ln_ab, hn_bf);

  // 3. QKVG projections (WMMA)
  dim3 gproj(DMODEL / 64, BN / 128);
  gemm_bf16_kernel<<<gproj, 256, 0, stream>>>(hn_bf, wqT, nullptr, Qf, BN, DMODEL, DMODEL);
  gemm_bf16_kernel<<<gproj, 256, 0, stream>>>(hn_bf, wkT, nullptr, Kf, BN, DMODEL, DMODEL);
  gemm_bf16_kernel<<<gproj, 256, 0, stream>>>(hn_bf, wvT, nullptr, Vf, BN, DMODEL, DMODEL);
  gemm_bf16_kernel<<<gproj, 256, 0, stream>>>(hn_bf, wgT, nullptr, Gf, BN, DMODEL, DMODEL);

  // 4. per-head LN + head-major layouts
  headprep_kernel<<<BHN / 8, 256, 0, stream>>>(Qf, Kf, Vf, Qln, Kln, Vh);

  // 5. cost matrix -> log kernel (WMMA content + fused cdist/geo bias)
  dim3 gcont(NTOK / 64, NTOK / 128, BH);
  content_kernel<<<gcont, 256, 0, stream>>>(Qln, Kln, wrel, x_res, wdist, eps, LK);

  // 6. Sinkhorn init + 20 iterations (u then v, matching the scan order)
  zero_kernel<<<(BHN + 255) / 256, 256, 0, stream>>>(log_u, BHN);
  zero_kernel<<<(BHN + 255) / 256, 256, 0, stream>>>(log_v, BHN);
  for (int it = 0; it < 20; ++it) {
    sinkhorn_u_kernel<<<BHN / 8, 256, 0, stream>>>(LK, log_v, mu, eps, log_u);
    sinkhorn_v_kernel<<<BH, 1024, 0, stream>>>(LK, log_u, nu, eps, log_v);
  }

  // 7. transport-weighted output + centroids
  transport_kernel<<<BHN / 8, 256, 0, stream>>>(LK, log_u, log_v, Vh, Gf, x_res, o_bf, xcent);

  // 8. output projection + residual: h2 = h + o @ wo
  gemm_bf16_kernel<<<gproj, 256, 0, stream>>>(o_bf, woT, h, h2, BN, DMODEL, DMODEL);

  // 9. FFN layernorm -> bf16
  ln512_kernel<<<BN / 8, 256, 0, stream>>>(h2, ln_fs, ln_fb, hff_bf);

  // 10. fused SwiGLU up-projection (dual WMMA GEMM)
  dim3 gff(DFF / 64, BN / 128);
  ffn_dual_kernel<<<gff, 256, 0, stream>>>(hff_bf, w1T, w2T, g_bf, BN, DFF, DMODEL);

  // 11. down projection + residual -> final h (written straight into d_out)
  gemm_bf16_kernel<<<gproj, 256, 0, stream>>>(g_bf, w3T, h2, out_h, BN, DMODEL, DFF);

  // 12. coordinate update output
  xout_kernel<<<(BN * 3 + 255) / 256, 256, 0, stream>>>(x_res, gamma, xcent, out_x);

  // 13. final Sinkhorn potentials to output
  hipMemcpyAsync(out_lu, log_u, (size_t)BHN * 4, hipMemcpyDeviceToDevice, stream);
  hipMemcpyAsync(out_lv, log_v, (size_t)BHN * 4, hipMemcpyDeviceToDevice, stream);
}
